// DeepONetsWithODEBranch_39487929319470
// MI455X (gfx1250) — matrix operand order repaired and checked by reading, not compiled, not run on hardware
//
#include <hip/hip_runtime.h>
#include <hip/hip_bf16.h>

typedef __attribute__((ext_vector_type(16))) _Float16 v16h;
typedef __attribute__((ext_vector_type(8)))  _Float16 v8h;
typedef __attribute__((ext_vector_type(8)))  float    v8f;

union V16 { v16h v; v8h h[2]; };

#if __has_builtin(__builtin_amdgcn_tanhf)
__device__ __forceinline__ float fast_tanh(float x) { return __builtin_amdgcn_tanhf(x); }
#else
__device__ __forceinline__ float fast_tanh(float x) { return tanhf(x); }
#endif

// ---------------------------------------------------------------------------
// A-tile (16x32 f16, M x K) from LDS, row-major, row stride == K.
// Lane m (0-15): rows M=m, K chunks [k0, k0+8) and [k0+16, k0+24).
// Lanes 16-31: same rows, chunks offset by +8.
// ---------------------------------------------------------------------------
__device__ __forceinline__ v16h load_a_tile(const _Float16* a, int ldA,
                                            int m0, int k0, int lane) {
  int row   = m0 + (lane & 15);
  int kbase = k0 + ((lane >> 4) << 3);
  const _Float16* p = a + row * ldA + kbase;
  V16 u;
  u.h[0] = *(const v8h*)(p);
  u.h[1] = *(const v8h*)(p + 16);
  return u.v;
}

// ---------------------------------------------------------------------------
// B-tile (32x16 f16, K x N) from global, weights stored TRANSPOSED [N][K].
// Lane n (0-15): column N=n0+n, K chunks [k0, k0+8) and [k0+16, k0+24); lanes 16-31: chunks offset by +8.
// Each lane: two 16-byte chunks 32 bytes apart -> two global_load_b128.
// ---------------------------------------------------------------------------
__device__ __forceinline__ v16h load_b_tile(const _Float16* wt, int K,
                                            int n0, int k0, int lane) {
  int col   = n0 + (lane & 15);
  int kbase = k0 + ((lane >> 4) << 3);
  const _Float16* p = wt + col * K + kbase;
  V16 u;
  u.h[0] = *(const v8h*)(p);
  u.h[1] = *(const v8h*)(p + 16);
  return u.v;
}

// ---------------------------------------------------------------------------
// WGP-scope prefetch via inline asm. __builtin_prefetch only reaches
// SE/SYS scope, which per the ISA table does NOT warm the WGP$ (it fills GL2 only).
// Scope field 0 (default, no suffix) = WGP:
// fills all cache levels on a miss. Addresses are always valid
// workspace bytes, so non-speculative WGP prefetch is safe.
// ---------------------------------------------------------------------------
__device__ __forceinline__ void prefetch_wgp(const void* p) {
  asm volatile("global_prefetch_b8 %0, off" :: "v"(p) : "memory");
}

// Prefetch the 16-column B slab (rows n0..n0+15 of Wt = 16*K*2 contiguous
// bytes) for a FUTURE layer into the WGP$. Fire-and-forget: no counters;
// overlaps the inter-layer barrier wait. K is compile-time at every call
// site, so the loop fully unrolls into a clause of prefetches.
__device__ __forceinline__ void prefetch_b_slab(const _Float16* wt, int K,
                                                int nt, int lane) {
  const char* base = (const char*)(wt + (nt << 4) * K) + lane * 128;
  const int bytes = K * 32;  // 16 rows * K * 2B
#pragma unroll
  for (int off = 0; off < bytes - 32 * 128 + 1; off += 32 * 128)
    prefetch_wgp(base + off);
}

#define ACT_TANH 0
#define ACT_RELU 1
#define ACT_NONE 2

// ---------------------------------------------------------------------------
// One MLP layer: out = act(A[MT*16 x K] @ W[K x N] + bias)
// Each wave owns whole N-tile columns; B tile loaded ONCE per k-step and
// reused across MT m-tiles (MT accumulators in registers).
// Loop bounds are scalar (wave id via readfirstlane) so EXEC stays all-1s
// through every WMMA, as the ISA requires.
// ---------------------------------------------------------------------------
template <int ACT, bool F16OUT, int MT>
__device__ __forceinline__ void wmma_layer(
    const _Float16* aBuf, const _Float16* wt, const float* bias,
    int K, int N, _Float16* outH, float* outF, int ldOut,
    int wave, int nWaves, int lane) {
  const int ntiles = N >> 4;
  for (int nt = wave; nt < ntiles; nt += nWaves) {
    const int n0 = nt << 4;
    v8f c[MT];
#pragma unroll
    for (int m = 0; m < MT; ++m) {
      v8f z = {0.f, 0.f, 0.f, 0.f, 0.f, 0.f, 0.f, 0.f};
      c[m] = z;
    }
    for (int k0 = 0; k0 < K; k0 += 32) {
      v16h b = load_b_tile(wt, K, n0, k0, lane);
#pragma unroll
      for (int m = 0; m < MT; ++m) {
        v16h a = load_a_tile(aBuf, K, m << 4, k0, lane);
        c[m] = __builtin_amdgcn_wmma_f32_16x16x32_f16(
            false, a, false, b, (short)0, c[m], false, false);
      }
    }
    const int   colg = n0 + (lane & 15);
    const float bv   = bias[colg];
#pragma unroll
    for (int m = 0; m < MT; ++m) {
      const int mbase = (m << 4) + ((lane >> 4) << 3);
#pragma unroll
      for (int j = 0; j < 8; ++j) {
        float v = c[m][j] + bv;
        if (ACT == ACT_TANH) v = fast_tanh(v);
        else if (ACT == ACT_RELU) v = fmaxf(v, 0.f);
        if (F16OUT) outH[(mbase + j) * ldOut + colg] = (_Float16)v;
        else        outF[(mbase + j) * ldOut + colg] = v;
      }
    }
  }
}

// ---------------------------------------------------------------------------
// Prep: convert f32 weight [K][N] -> transposed f16 [N][K]
// ---------------------------------------------------------------------------
__global__ void transpose_to_f16(const float* __restrict__ src,
                                 _Float16* __restrict__ dst, int K, int N) {
  int idx = blockIdx.x * blockDim.x + threadIdx.x;
  if (idx < K * N) {
    int k = idx % K, n = idx / K;
    dst[n * K + k] = (_Float16)src[k * N + n];
  }
}

// ---------------------------------------------------------------------------
// Branch ODE: 4 WGs x 512 threads (16 waves). Each WG owns 16 batch rows
// (one M-tile) and runs all 49 RK45 steps (6 drift MLP evals each).
// State x and stage slopes k[6] live in registers (4 f32 per thread);
// activations ping-pong through LDS f16; layer-4 output via LDS f32.
// Next-layer weight slabs are prefetched (WGP scope) before each barrier to
// hide L2 latency on the serial layer chain.
// ---------------------------------------------------------------------------
__global__ __launch_bounds__(512)
void branch_ode_kernel(const float* __restrict__ parameter,
                       const _Float16* __restrict__ w1t,
                       const _Float16* __restrict__ w2t,
                       const _Float16* __restrict__ w3t,
                       const _Float16* __restrict__ w4t,
                       const float* __restrict__ b1, const float* __restrict__ b2,
                       const float* __restrict__ b3, const float* __restrict__ b4,
                       float* __restrict__ branch_out) {
  extern __shared__ char smem[];
  _Float16* bufA = (_Float16*)smem;                    // 16x512 f16 (16 KB)
  _Float16* bufB = (_Float16*)(smem + 16 * 512 * 2);   // 16x512 f16 (16 KB)
  float*    kBuf = (float*)(smem + 2 * 16 * 512 * 2);  // 16x128 f32 ( 8 KB)

  const int t    = threadIdx.x;
  const int wave = __builtin_amdgcn_readfirstlane(t >> 5);  // scalar wave id
  const int lane = t & 31;
  const int nWaves = 16;
  const int wg = blockIdx.x;
  const float dt = 1.0f / 49.0f;

  const int flat = t * 4;                 // this thread's 4 contiguous state elems
  const int lm   = flat >> 7;             // local row 0..15
  const int col  = flat & 127;
  const int grow = (wg << 4) + lm;        // global batch row

  float x[4];
  {
    const float4 x4 = *(const float4*)(parameter + grow * 128 + col);
    x[0] = x4.x; x[1] = x4.y; x[2] = x4.z; x[3] = x4.w;
  }
  float ks[6][4];

  const float Arow[5][5] = {
      {1.f/5.f, 0.f, 0.f, 0.f, 0.f},
      {3.f/40.f, 9.f/40.f, 0.f, 0.f, 0.f},
      {44.f/45.f, -56.f/15.f, 32.f/9.f, 0.f, 0.f},
      {19372.f/6561.f, -25360.f/2187.f, 64448.f/6561.f, -212.f/729.f, 0.f},
      {9017.f/3168.f, -355.f/33.f, 46732.f/5247.f, 49.f/176.f, -5103.f/18656.f}};
  const float Bw[6] = {35.f/384.f, 0.f, 500.f/1113.f, 125.f/192.f,
                       -2187.f/6784.f, 11.f/84.f};

  // warm L1 weights before first use
  prefetch_b_slab(w1t, 128, wave, lane);

  for (int step = 0; step < 49; ++step) {
#pragma unroll
    for (int s = 0; s < 6; ++s) {
      // stage input xs = x + dt * sum_j A[s-1][j] * k[j]  (registers)
#pragma unroll
      for (int i = 0; i < 4; ++i) {
        float acc = x[i];
        if (s > 0) {
          float inc = 0.f;
#pragma unroll
          for (int j = 0; j < 5; ++j)
            if (j < s) inc += Arow[s - 1][j] * ks[j][i];
          acc += dt * inc;
        }
        bufA[flat + i] = (_Float16)acc;   // [16][128] f16, row stride 128
      }
      __syncthreads();
      wmma_layer<ACT_TANH, true , 1>(bufA, w1t, b1, 128, 256, bufB, nullptr, 256, wave, nWaves, lane);
      prefetch_b_slab(w2t, 256, wave, lane);         // L2 tiles: nt = wave, wave+16
      prefetch_b_slab(w2t, 256, wave + 16, lane);
      __syncthreads();
      wmma_layer<ACT_TANH, true , 1>(bufB, w2t, b2, 256, 512, bufA, nullptr, 512, wave, nWaves, lane);
      prefetch_b_slab(w3t, 512, wave, lane);         // L3 tiles: nt = wave
      __syncthreads();
      wmma_layer<ACT_TANH, true , 1>(bufA, w3t, b3, 512, 256, bufB, nullptr, 256, wave, nWaves, lane);
      if (wave < 8) prefetch_b_slab(w4t, 256, wave, lane);  // L4 tiles: nt = wave (<8)
      __syncthreads();
      wmma_layer<ACT_NONE, false, 1>(bufB, w4t, b4, 256, 128, nullptr, kBuf, 128, wave, nWaves, lane);
      prefetch_b_slab(w1t, 128, wave, lane);         // next stage's L1
      __syncthreads();
#pragma unroll
      for (int i = 0; i < 4; ++i) ks[s][i] = kBuf[flat + i];
    }
#pragma unroll
    for (int i = 0; i < 4; ++i) {
      float inc = 0.f;
#pragma unroll
      for (int j = 0; j < 6; ++j) inc += Bw[j] * ks[j][i];
      x[i] += dt * inc;
    }
  }
  float4 o; o.x = x[0]; o.y = x[1]; o.z = x[2]; o.w = x[3];
  *(float4*)(branch_out + grow * 128 + col) = o;
}

// ---------------------------------------------------------------------------
// Trunk + combine: 2048 WGs x 256 threads (8 waves). Each WG: 64 coordinate
// rows of one batch (4 m-tiles; B tile loaded once per k-step, reused 4x ->
// halves L2 weight traffic vs 2 m-tiles). L1 (K=4) in VALU, L2/L3 via WMMA,
// then fused out = h3 . (branch*oW) + ob.  LDS: 128 KB/WG (2 WGs/WGP).
// ---------------------------------------------------------------------------
__global__ __launch_bounds__(256)
void trunk_kernel(const float* __restrict__ coords,
                  const float* __restrict__ tW1, const float* __restrict__ tb1,
                  const _Float16* __restrict__ w2t, const float* __restrict__ tb2,
                  const _Float16* __restrict__ w3t, const float* __restrict__ tb3,
                  const float* __restrict__ oW, const float* __restrict__ ob,
                  const float* __restrict__ branch_out,
                  float* __restrict__ out) {
  extern __shared__ char smem[];
  _Float16* buf1 = (_Float16*)smem;             // 64x512 f16 (64 KB)
  _Float16* buf2 = (_Float16*)(smem + 65536);   // 64x512 f16 (64 KB)
  float*    cbuf = (float*)(smem + 65536);      // 64x4 f32, aliases buf2 (pre-L2)
  float*    h3   = (float*)smem;                // 64x128 f32, aliases buf1 (post-L2)

  const int t    = threadIdx.x;
  const int wave = __builtin_amdgcn_readfirstlane(t >> 5);  // scalar wave id
  const int lane = t & 31;
  const int nWaves = 8;
  const int b  = blockIdx.x >> 5;               // 32 row-blocks per batch
  const int p0 = (blockIdx.x & 31) << 6;        // 64 rows per WG

  if (t < 256) cbuf[t] = coords[(b * 2048 + p0) * 4 + t];  // 64 rows x 4, contiguous
  __syncthreads();

  // L1: [64x4] @ [4x512] + relu -> buf1 f16 (ld 512); K=4 too small for WMMA
  for (int idx = t; idx < 64 * 512; idx += 256) {
    const int r = idx >> 9, n = idx & 511;
    float acc = tb1[n];
#pragma unroll
    for (int k = 0; k < 4; ++k) acc += cbuf[r * 4 + k] * tW1[k * 512 + n];
    buf1[idx] = (_Float16)fmaxf(acc, 0.f);
  }
  __syncthreads();
  wmma_layer<ACT_RELU, true , 4>(buf1, w2t, tb2, 512, 512, buf2, nullptr, 512, wave, nWaves, lane);
  __syncthreads();
  wmma_layer<ACT_RELU, false, 4>(buf2, w3t, tb3, 512, 128, nullptr, h3, 128, wave, nWaves, lane);
  __syncthreads();

  // combine: out[b, p0+r] = sum_k h3[r][k] * branch[b][k] * oW[k] + ob
  if (t < 64) {
    float acc = ob[0];
    for (int k = 0; k < 128; ++k)
      acc += h3[t * 128 + k] * branch_out[b * 128 + k] * oW[k];
    out[b * 2048 + p0 + t] = acc;
  }
}

// ---------------------------------------------------------------------------
// Workspace layout (bytes)
// ---------------------------------------------------------------------------
#define OFF_BW1T 0u         // [256][128] f16 ->  65536
#define OFF_BW2T 65536u     // [512][256] f16 -> 262144
#define OFF_BW3T 327680u    // [256][512] f16 -> 262144
#define OFF_BW4T 589824u    // [128][256] f16 ->  65536
#define OFF_TW2T 655360u    // [512][512] f16 -> 524288
#define OFF_TW3T 1179648u   // [128][512] f16 -> 131072
#define OFF_BOUT 1310720u   // [64][128]  f32 ->  32768

extern "C" void kernel_launch(void* const* d_in, const int* in_sizes, int n_in,
                              void* d_out, int out_size, void* d_ws, size_t ws_size,
                              hipStream_t stream) {
  const float* parameter = (const float*)d_in[0];
  const float* coords    = (const float*)d_in[1];
  const float* bW1 = (const float*)d_in[2];  const float* bb1 = (const float*)d_in[3];
  const float* bW2 = (const float*)d_in[4];  const float* bb2 = (const float*)d_in[5];
  const float* bW3 = (const float*)d_in[6];  const float* bb3 = (const float*)d_in[7];
  const float* bW4 = (const float*)d_in[8];  const float* bb4 = (const float*)d_in[9];
  const float* tW1 = (const float*)d_in[10]; const float* tb1 = (const float*)d_in[11];
  const float* tW2 = (const float*)d_in[12]; const float* tb2 = (const float*)d_in[13];
  const float* tW3 = (const float*)d_in[14]; const float* tb3 = (const float*)d_in[15];
  const float* oW  = (const float*)d_in[16]; const float* ob  = (const float*)d_in[17];
  float* out = (float*)d_out;

  char* ws = (char*)d_ws;
  _Float16* bW1t = (_Float16*)(ws + OFF_BW1T);
  _Float16* bW2t = (_Float16*)(ws + OFF_BW2T);
  _Float16* bW3t = (_Float16*)(ws + OFF_BW3T);
  _Float16* bW4t = (_Float16*)(ws + OFF_BW4T);
  _Float16* tW2t = (_Float16*)(ws + OFF_TW2T);
  _Float16* tW3t = (_Float16*)(ws + OFF_TW3T);
  float*    bout = (float*)(ws + OFF_BOUT);

  // Prep: transpose+convert weights to f16 [N][K]
  {
    struct { const float* s; _Float16* d; int K, N; } jobs[6] = {
        {bW1, bW1t, 128, 256}, {bW2, bW2t, 256, 512},
        {bW3, bW3t, 512, 256}, {bW4, bW4t, 256, 128},
        {tW2, tW2t, 512, 512}, {tW3, tW3t, 512, 128}};
    for (int i = 0; i < 6; ++i) {
      int total = jobs[i].K * jobs[i].N;
      transpose_to_f16<<<(total + 255) / 256, 256, 0, stream>>>(
          jobs[i].s, jobs[i].d, jobs[i].K, jobs[i].N);
    }
  }

  // Branch ODE: 4 WGs x 512 threads, 40 KB dynamic LDS each
  branch_ode_kernel<<<4, 512, 40960, stream>>>(
      parameter, bW1t, bW2t, bW3t, bW4t, bb1, bb2, bb3, bb4, bout);

  // Trunk + combine: 64 batches x 32 row-blocks, 128 KB dynamic LDS
  trunk_kernel<<<64 * 32, 256, 131072, stream>>>(
      coords, tW1, tb1, tW2t, tb2, tW3t, tb3, oW, ob, bout, out);
}
